// AdaptiveAttention_50964081934580
// MI455X (gfx1250) — compile-verified, matrix operand order checked by reading, not run on hardware
//
#include <hip/hip_runtime.h>
#include <hip/hip_bf16.h>

// ---------------------------------------------------------------------------
// AdaptiveAttention for MI455X (gfx1250): bf16 WMMA (f32 accumulate) everywhere.
// Memory floor ~28us (537MB attn_weights write dominates); bf16 16x16x32 WMMA
// keeps all GEMMs below that floor, whereas f32 16x16x4 WMMA would not.
// ---------------------------------------------------------------------------

typedef __attribute__((ext_vector_type(16))) __bf16 v16bf;
typedef __attribute__((ext_vector_type(8)))  float  v8f;

#define BB 2
#define SS 2048
#define HSZ 1024
#define NHD 16
#define NPAT 8
#define HDIM 64
#define ATT_SCALE 0.125f  // 1/sqrt(64)

// K-dim element mapping for 16-bit A/B fragments (ISA 7.12.2, 16x32 bf16):
// lanes 0-15: elems 0..7 -> K 0..7,  elems 8..15 -> K 16..23
// lanes16-31: elems 0..7 -> K 8..15, elems 8..15 -> K 24..31
__device__ __forceinline__ int kmap_e(int lane, int e) {
  const int hi = (lane >> 4) & 1;
  const int base = (e < 8) ? (hi ? 8 : 0) : (hi ? 24 : 16);
  return base + (e & 7);
}

__device__ __forceinline__ v8f wmma_bf16(v16bf a, v16bf b, v8f c) {
  return __builtin_amdgcn_wmma_f32_16x16x32_bf16(false, a, false, b,
                                                 (short)0, c, false, false);
}

__device__ __forceinline__ float half16_max(float v) {
  v = fmaxf(v, __shfl_xor(v, 1, 32));
  v = fmaxf(v, __shfl_xor(v, 2, 32));
  v = fmaxf(v, __shfl_xor(v, 4, 32));
  v = fmaxf(v, __shfl_xor(v, 8, 32));
  return v;
}
__device__ __forceinline__ float half16_sum(float v) {
  v += __shfl_xor(v, 1, 32);
  v += __shfl_xor(v, 2, 32);
  v += __shfl_xor(v, 4, 32);
  v += __shfl_xor(v, 8, 32);
  return v;
}

// ---------------------------------------------------------------------------
// Pattern selector (tiny)
// ---------------------------------------------------------------------------
__global__ void aa_mean_pool(const float* __restrict__ q, float* __restrict__ mean) {
  const int c = blockIdx.x * blockDim.x + threadIdx.x;   // [0, BB*HSZ)
  const int b = c / HSZ, col = c % HSZ;
  float s = 0.f;
  for (int i = 0; i < SS; ++i) s += q[((size_t)b * SS + i) * HSZ + col];
  mean[c] = s * (1.f / (float)SS);
}

__global__ void aa_pattern_selector(const float* __restrict__ mean,
                                    const float* __restrict__ Wp1, const float* __restrict__ bp1,
                                    const float* __restrict__ Wp2, const float* __restrict__ bp2,
                                    const float* __restrict__ patterns,
                                    float* __restrict__ pat) {
  __shared__ float hbuf[HSZ / 2];
  __shared__ float pbuf[NPAT];
  const int b = blockIdx.x, t = threadIdx.x;
  const float* mb = mean + (size_t)b * HSZ;
  for (int j = t; j < HSZ / 2; j += blockDim.x) {
    float s = bp1[j];
    for (int i = 0; i < HSZ; ++i) s += mb[i] * Wp1[(size_t)i * (HSZ / 2) + j];
    hbuf[j] = fmaxf(s, 0.f);  // relu
  }
  __syncthreads();
  if (t < NPAT) {
    float s = bp2[t];
    for (int i = 0; i < HSZ / 2; ++i) s += hbuf[i] * Wp2[(size_t)i * NPAT + t];
    pbuf[t] = s;
  }
  __syncthreads();
  if (t == 0) {
    float m = pbuf[0];
    for (int p = 1; p < NPAT; ++p) m = fmaxf(m, pbuf[p]);
    float sum = 0.f;
    for (int p = 0; p < NPAT; ++p) { pbuf[p] = __expf(pbuf[p] - m); sum += pbuf[p]; }
    const float inv = 1.f / sum;
    for (int p = 0; p < NPAT; ++p) pbuf[p] *= inv;
  }
  __syncthreads();
  if (t < NHD) {
    float s = 0.f;
    for (int p = 0; p < NPAT; ++p) s += pbuf[p] * patterns[p * NHD + t];
    pat[b * NHD + t] = s;
  }
}

// ---------------------------------------------------------------------------
// Generic GEMM: Out[M,N] = X[M,K] @ W[K,N] + bias, bf16 WMMA, f32 accumulate.
// Block tile 128x128, K-step 32, 8 waves; each wave does a 32x64 sub-tile
// (2x4 v8f accumulators). XF32: X is fp32 (converted to bf16 in LDS).
// HEADOUT: write bf16 in [B,NH,S,HD] head-major layout; else fp32 row-major.
// ---------------------------------------------------------------------------
template <bool XF32, bool HEADOUT>
__global__ __launch_bounds__(256) void aa_gemm_bf16_wmma(
    const void* __restrict__ Xv, const float* __restrict__ W,
    const float* __restrict__ bias, void* __restrict__ Out,
    int M, int N, int K) {
  __shared__ __bf16 As[128][40];    // +8 pad: conflict-free column reads
  __shared__ __bf16 Bs[32][136];
  const int tid  = threadIdx.x;
  const int lane = tid & 31;
  const int wid  = tid >> 5;
  const int wm   = wid & 3;          // 4 waves along M (32 rows each)
  const int wn   = wid >> 2;         // 2 waves along N (64 cols each)
  const int lm   = lane & 15;
  const int hi8  = ((lane >> 4) & 1) << 3;
  const int bm = blockIdx.x * 128;
  const int bn = blockIdx.y * 128;
  const float*  Xf = (const float*)Xv;
  const __bf16* Xb = (const __bf16*)Xv;

  v8f acc[2][4];
#pragma unroll
  for (int mi = 0; mi < 2; ++mi)
#pragma unroll
    for (int ni = 0; ni < 4; ++ni)
#pragma unroll
      for (int r = 0; r < 8; ++r) acc[mi][ni][r] = 0.f;

  const int ar = tid >> 1, ac = (tid & 1) * 16;   // 128x32 A tile: 16 elems/thread
  const int br = tid >> 3, bc = (tid & 7) * 16;   // 32x128 B tile: 16 elems/thread

  for (int k0 = 0; k0 < K; k0 += 32) {
    const size_t abase = (size_t)(bm + ar) * K + k0 + ac;
#pragma unroll
    for (int i = 0; i < 16; ++i) {
      const float v = XF32 ? Xf[abase + i] : (float)Xb[abase + i];
      As[ar][ac + i] = (__bf16)v;
    }
    const size_t bbase = (size_t)(k0 + br) * N + bn + bc;
#pragma unroll
    for (int i = 0; i < 16; ++i) Bs[br][bc + i] = (__bf16)W[bbase + i];
    if (k0 + 32 < K)  // global_prefetch_b8 for next W tile
      __builtin_prefetch(&W[(size_t)(k0 + 32 + br) * N + bn + bc], 0, 1);
    __syncthreads();

    v16bf afr[2], bfr[4];
#pragma unroll
    for (int mi = 0; mi < 2; ++mi) {
      const int Mr = wm * 32 + mi * 16 + lm;
#pragma unroll
      for (int e = 0; e < 16; ++e) afr[mi][e] = As[Mr][kmap_e(lane, e)];
    }
#pragma unroll
    for (int ni = 0; ni < 4; ++ni) {
      const int Nc = wn * 64 + ni * 16 + lm;
#pragma unroll
      for (int e = 0; e < 16; ++e) bfr[ni][e] = Bs[kmap_e(lane, e)][Nc];
    }
#pragma unroll
    for (int mi = 0; mi < 2; ++mi)
#pragma unroll
      for (int ni = 0; ni < 4; ++ni)
        acc[mi][ni] = wmma_bf16(afr[mi], bfr[ni], acc[mi][ni]);
    __syncthreads();
  }

#pragma unroll
  for (int mi = 0; mi < 2; ++mi)
#pragma unroll
    for (int ni = 0; ni < 4; ++ni) {
      const int col = bn + wn * 64 + ni * 16 + lm;
      const float bv = bias[col];
#pragma unroll
      for (int r = 0; r < 8; ++r) {
        const int row = bm + wm * 32 + mi * 16 + r + hi8;
        const float v = acc[mi][ni][r] + bv;
        if (HEADOUT) {
          const int b = row / SS, s = row % SS;
          const int h = col / HDIM, hd = col % HDIM;
          ((__bf16*)Out)[(((size_t)b * NHD + h) * SS + s) * HDIM + hd] = (__bf16)v;
        } else {
          ((float*)Out)[(size_t)row * N + col] = v;
        }
      }
    }
}

// ---------------------------------------------------------------------------
// Fused attention: one block per (b, h, 16-query strip).
// Each wave owns 256 key columns: 16 v8f score accumulators live in VGPRs for
// the full 16x2048 strip. Softmax via shfl_xor (16-lane halves) + cross-wave
// LDS reduction. Normalized weights -> d_out (fp32), then reused as bf16
// A-fragments (per-wave LDS transpose stage) for P@V; wave partials combined
// with ds_add_f32 LDS atomics.
// ---------------------------------------------------------------------------
__global__ __launch_bounds__(256) void aa_attn_softmax_ctx(
    const __bf16* __restrict__ Qb, const __bf16* __restrict__ Kb,
    const __bf16* __restrict__ Vb, const int* __restrict__ mask,
    const float* __restrict__ pat, float* __restrict__ attn_out,
    __bf16* __restrict__ ctx_out) {
  __shared__ float  red[8][16];
  __shared__ __bf16 Pstage[8][16][32];
  __shared__ float  ctxLds[16][64];

  const int tid = threadIdx.x, lane = tid & 31, w = tid >> 5;
  const int lm  = lane & 15;
  const int hi8 = ((lane >> 4) & 1) << 3;
  const int strips = SS / 16;
  const int q0 = (blockIdx.x % strips) * 16;
  const int bh = blockIdx.x / strips;
  const int b = bh / NHD, h = bh % NHD;
  const size_t bhS = (size_t)bh * SS;
  const float pm = ATT_SCALE * pat[bh];
  const int colBase = w * 256;

  for (int i = tid; i < 16 * 64; i += 256) ((float*)ctxLds)[i] = 0.f;

  // Q A-fragments (16 rows x 64 K, two K=32 fragments)
  v16bf qa[2];
#pragma unroll
  for (int kf = 0; kf < 2; ++kf)
#pragma unroll
    for (int e = 0; e < 16; ++e)
      qa[kf][e] = Qb[(bhS + q0 + lm) * HDIM + kf * 32 + kmap_e(lane, e)];

  // scores: 16x256 per wave, S = Q (16x64) @ K^T (64x256)
  v8f accS[16];
#pragma unroll
  for (int t = 0; t < 16; ++t)
#pragma unroll
    for (int r = 0; r < 8; ++r) accS[t][r] = 0.f;

#pragma unroll
  for (int t = 0; t < 16; ++t) {
    const int n0 = colBase + t * 16;
#pragma unroll
    for (int kf = 0; kf < 2; ++kf) {
      v16bf kb;  // B = K^T : B[kd][n] = K[n0+n][kd]
#pragma unroll
      for (int e = 0; e < 16; ++e)
        kb[e] = Kb[(bhS + n0 + lm) * HDIM + kf * 32 + kmap_e(lane, e)];
      accS[t] = wmma_bf16(qa[kf], kb, accS[t]);
    }
  }

  // scale * pattern, mask (mask==0 -> -1e9)
#pragma unroll
  for (int t = 0; t < 16; ++t)
#pragma unroll
    for (int r = 0; r < 8; ++r) {
      const int row = q0 + r + hi8;
      const int col = colBase + t * 16 + lm;
      float sc = accS[t][r] * pm;
      if (mask[((size_t)b * SS + row) * SS + col] == 0) sc = -1e9f;
      accS[t][r] = sc;
    }

  // row max (wave-local shuffle, then cross-wave via LDS)
  float fmax_[8];
  {
    float rl[8];
#pragma unroll
    for (int r = 0; r < 8; ++r) {
      float m = accS[0][r];
#pragma unroll
      for (int t = 1; t < 16; ++t) m = fmaxf(m, accS[t][r]);
      rl[r] = half16_max(m);
    }
    if (lm == 0) {
#pragma unroll
      for (int r = 0; r < 8; ++r) red[w][hi8 + r] = rl[r];
    }
    __syncthreads();
#pragma unroll
    for (int r = 0; r < 8; ++r) {
      float m = red[0][hi8 + r];
#pragma unroll
      for (int ww = 1; ww < 8; ++ww) m = fmaxf(m, red[ww][hi8 + r]);
      fmax_[r] = m;
    }
    __syncthreads();
  }

  // exp + row sum
  float fsum[8];
  {
    float rl[8];
#pragma unroll
    for (int r = 0; r < 8; ++r) {
      float s = 0.f;
#pragma unroll
      for (int t = 0; t < 16; ++t) {
        const float p = __expf(accS[t][r] - fmax_[r]);
        accS[t][r] = p;
        s += p;
      }
      rl[r] = half16_sum(s);
    }
    if (lm == 0) {
#pragma unroll
      for (int r = 0; r < 8; ++r) red[w][hi8 + r] = rl[r];
    }
    __syncthreads();
#pragma unroll
    for (int r = 0; r < 8; ++r) {
      float s = 0.f;
#pragma unroll
      for (int ww = 0; ww < 8; ++ww) s += red[ww][hi8 + r];
      fsum[r] = 1.f / s;
    }
  }

  // normalize + write attention weights (fp32, [B,NH,S,S])
  float* aout = attn_out + (bhS + q0) * SS;
#pragma unroll
  for (int t = 0; t < 16; ++t)
#pragma unroll
    for (int r = 0; r < 8; ++r) {
      const float p = accS[t][r] * fsum[r];
      accS[t][r] = p;
      aout[(size_t)(r + hi8) * SS + colBase + t * 16 + lm] = p;
    }

  // context: C(16x64) = P(16x2048) @ V(2048x64); wave handles its 256 keys
  v8f accC[4];
#pragma unroll
  for (int nf = 0; nf < 4; ++nf)
#pragma unroll
    for (int r = 0; r < 8; ++r) accC[nf][r] = 0.f;

#pragma unroll
  for (int kk = 0; kk < 8; ++kk) {
    // stage C-layout 16x32 probs into LDS, re-read in A-fragment layout
#pragma unroll
    for (int half = 0; half < 2; ++half) {
      const int t = kk * 2 + half;
#pragma unroll
      for (int r = 0; r < 8; ++r)
        Pstage[w][r + hi8][half * 16 + lm] = (__bf16)accS[t][r];
    }
    __syncthreads();
    v16bf pa;
#pragma unroll
    for (int e = 0; e < 16; ++e) pa[e] = Pstage[w][lm][kmap_e(lane, e)];
#pragma unroll
    for (int nf = 0; nf < 4; ++nf) {
      v16bf vbf;  // B = V chunk: B[k][n] = V[key][nf*16+n]
#pragma unroll
      for (int e = 0; e < 16; ++e) {
        const int key = colBase + kk * 32 + kmap_e(lane, e);
        vbf[e] = Vb[(bhS + key) * HDIM + nf * 16 + lm];
      }
      accC[nf] = wmma_bf16(pa, vbf, accC[nf]);
    }
    __syncthreads();
  }

  // combine wave partials (ds_add_f32) and emit bf16 context [B,S,HS]
#pragma unroll
  for (int nf = 0; nf < 4; ++nf)
#pragma unroll
    for (int r = 0; r < 8; ++r)
      atomicAdd(&ctxLds[r + hi8][nf * 16 + lm], accC[nf][r]);
  __syncthreads();
  for (int i = tid; i < 16 * 64; i += 256) {
    const int r = i >> 6, hd = i & 63;
    ctx_out[((size_t)b * SS + q0 + r) * HSZ + h * HDIM + hd] = (__bf16)ctxLds[r][hd];
  }
}

// ---------------------------------------------------------------------------
// Launcher
// ---------------------------------------------------------------------------
extern "C" void kernel_launch(void* const* d_in, const int* in_sizes, int n_in,
                              void* d_out, int out_size, void* d_ws, size_t ws_size,
                              hipStream_t stream) {
  (void)in_sizes; (void)n_in; (void)out_size; (void)ws_size;
  const float* query = (const float*)d_in[0];
  const float* key_  = (const float*)d_in[1];
  const float* value = (const float*)d_in[2];
  const int*   amask = (const int*)d_in[3];
  const float* Wq = (const float*)d_in[4];   const float* bq = (const float*)d_in[5];
  const float* Wk = (const float*)d_in[6];   const float* bk = (const float*)d_in[7];
  const float* Wv = (const float*)d_in[8];   const float* bv = (const float*)d_in[9];
  const float* Wo = (const float*)d_in[10];  const float* bo = (const float*)d_in[11];
  const float* Wp1 = (const float*)d_in[12]; const float* bp1 = (const float*)d_in[13];
  const float* Wp2 = (const float*)d_in[14]; const float* bp2 = (const float*)d_in[15];
  const float* patterns = (const float*)d_in[16];

  const size_t qkvElems = (size_t)BB * SS * HSZ;  // 4,194,304
  __bf16* Qbf = (__bf16*)d_ws;
  __bf16* Kbf = Qbf + qkvElems;
  __bf16* Vbf = Kbf + qkvElems;
  __bf16* Cbf = Vbf + qkvElems;
  float* meanp = (float*)(Cbf + qkvElems);
  float* patp  = meanp + BB * HSZ;

  float* out_main = (float*)d_out;              // [B,S,HS] fp32
  float* attn_out = out_main + qkvElems;        // [B,NH,S,S] fp32

  aa_mean_pool<<<(BB * HSZ) / 256, 256, 0, stream>>>(query, meanp);
  aa_pattern_selector<<<BB, 256, 0, stream>>>(meanp, Wp1, bp1, Wp2, bp2, patterns, patp);

  dim3 gp((BB * SS) / 128, HSZ / 128);  // 32 x 8 blocks
  aa_gemm_bf16_wmma<true, true><<<gp, 256, 0, stream>>>(query, Wq, bq, Qbf, BB * SS, HSZ, HSZ);
  aa_gemm_bf16_wmma<true, true><<<gp, 256, 0, stream>>>(key_,  Wk, bk, Kbf, BB * SS, HSZ, HSZ);
  aa_gemm_bf16_wmma<true, true><<<gp, 256, 0, stream>>>(value, Wv, bv, Vbf, BB * SS, HSZ, HSZ);

  aa_attn_softmax_ctx<<<BB * NHD * (SS / 16), 256, 0, stream>>>(
      Qbf, Kbf, Vbf, amask, patp, attn_out, Cbf);

  aa_gemm_bf16_wmma<false, false><<<gp, 256, 0, stream>>>(Cbf, Wo, bo, out_main, BB * SS, HSZ, HSZ);
}
